// dMaSIFSiteEmbed_1898375545075
// MI455X (gfx1250) — matrix operand order repaired and checked by reading, not compiled, not run on hardware
//
#include <hip/hip_runtime.h>
#include <hip/hip_bf16.h>

#define NPTS   3000
#define NPAD   3008           // 188 * 16 = 94 * 32
#define NTILE  188
#define NCHUNK (NPAD / 32)    // 94
#define EMB    128
#define CUTS   8
#define RADIUS 9.0f

typedef __attribute__((ext_vector_type(16))) _Float16 v16h;
typedef __attribute__((ext_vector_type(8)))  _Float16 v8h;
typedef __attribute__((ext_vector_type(8)))  float    v8f;
typedef unsigned int u32;
typedef __attribute__((ext_vector_type(4))) u32 u32x4;
typedef __attribute__((ext_vector_type(8))) int i32x8;
typedef __attribute__((ext_vector_type(4))) int i32x4;

#ifndef __has_builtin
#define __has_builtin(x) 0
#endif
#if __has_builtin(__builtin_amdgcn_tensor_load_to_lds) && \
    __has_builtin(__builtin_amdgcn_s_wait_tensorcnt)
#define HAVE_TDM 1
#else
#define HAVE_TDM 0
#endif

// ---------------------------------------------------------------------------
// input index map (jax tree_flatten: dict keys sorted alphabetically)
// ---------------------------------------------------------------------------
#define IN_FEATS 0
#define LIDX(l,k) (1 + (l)*22 + (k))
#define P_A1 0
#define P_A2 1
#define P_B1 2
#define P_B2 3
#define P_GNINB 4
#define P_GNING 5
#define P_GNOUTB 6
#define P_GNOUTG 7
#define P_INB1 8
#define P_INB2 9
#define P_INW1 10
#define P_INW2 11
#define P_LINB1 12
#define P_LINB2 13
#define P_LINW1 14
#define P_LINW2 15
#define P_OUTB1 16
#define P_OUTB2 17
#define P_OUTW1 18
#define P_OUTW2 19
#define P_SKIPB 20
#define P_SKIPW 21
#define IN_ORIB1 67
#define IN_ORIB2 68
#define IN_ORIW1 69
#define IN_ORIW2 70
#define IN_NRM 71
#define IN_XYZ 72

// ---------------------------------------------------------------------------
// tiny utility kernels
// ---------------------------------------------------------------------------
__global__ void k_zero(float* p, int n) {
    int i = blockIdx.x * blockDim.x + threadIdx.x;
    if (i < n) p[i] = 0.f;
}

__global__ void k_cvt16(const float* __restrict__ src, _Float16* __restrict__ dst, int n) {
    int i = blockIdx.x * blockDim.x + threadIdx.x;
    if (i < n) dst[i] = (_Float16)src[i];
}

// zero-pad features (3000 x 16) -> (3008 x 16)
__global__ void k_padfeats(const float* __restrict__ src, float* __restrict__ dst) {
    int idx = blockIdx.x * blockDim.x + threadIdx.x;
    if (idx >= NPAD * 16) return;
    int i = idx >> 4;
    dst[idx] = (i < NPTS) ? src[idx] : 0.f;
}

// ---------------------------------------------------------------------------
// orientation score MLP: 16 -> 16 (leaky) -> 1
// ---------------------------------------------------------------------------
__global__ void k_scores(const float* __restrict__ feats,
                         const float* __restrict__ w1, const float* __restrict__ b1,
                         const float* __restrict__ w2, const float* __restrict__ b2,
                         float* __restrict__ scores) {
    int i = blockIdx.x * blockDim.x + threadIdx.x;
    if (i >= NPTS) return;
    float f[16];
#pragma unroll
    for (int k = 0; k < 16; ++k) f[k] = feats[i * 16 + k];
    float sc = 0.f;
#pragma unroll
    for (int u = 0; u < 16; ++u) {
        float h = b1[u];
#pragma unroll
        for (int k = 0; k < 16; ++k) h = fmaf(w1[u * 16 + k], f[k], h);
        h = (h >= 0.f) ? h : 0.2f * h;
        sc = fmaf(w2[u], h, sc);
    }
    scores[i] = sc + b2[0];
}

// ---------------------------------------------------------------------------
// load_mesh: pairwise weighted mean -> tangent frame nuv (3000 x 3 x 3)
// ---------------------------------------------------------------------------
__global__ void k_mesh(const float* __restrict__ xyz, const float* __restrict__ nrm,
                       const float* __restrict__ scores, float* __restrict__ nuv) {
    int i = blockIdx.x * blockDim.x + threadIdx.x;
    if (i >= NPTS) return;
    const float invR = 1.0f / RADIUS;
    float pix = xyz[i * 3 + 0] * invR, piy = xyz[i * 3 + 1] * invR, piz = xyz[i * 3 + 2] * invR;
    float nx = nrm[i * 3 + 0], ny = nrm[i * 3 + 1], nz = nrm[i * 3 + 2];
    float s = (nz > 0.f) ? 1.f : -1.f;
    float a = -1.f / (s + nz);
    float b = nx * ny * a;
    float ux = 1.f + s * nx * nx * a, uy = s * b,            uz = -s * nx;
    float vx = b,                     vy = s + ny * ny * a,  vz = -ny;
    float Vx = 0.f, Vy = 0.f, Vz = 0.f, wsum = 0.f;
    for (int j = 0; j < NPTS; ++j) {
        float pjx = xyz[j * 3 + 0] * invR, pjy = xyz[j * 3 + 1] * invR, pjz = xyz[j * 3 + 2] * invR;
        float dx = pjx - pix, dy = pjy - piy, dz = pjz - piz;
        float d2 = dx * dx + dy * dy + dz * dz;
        float cs = nx * nrm[j * 3 + 0] + ny * nrm[j * 3 + 1] + nz * nrm[j * 3 + 2];
        float t = 2.f - cs;
        float w = __expf(-d2 * t * t) * scores[j];
        Vx = fmaf(w, pjx, Vx); Vy = fmaf(w, pjy, Vy); Vz = fmaf(w, pjz, Vz);
        wsum += w;
    }
    Vx -= wsum * pix; Vy -= wsum * piy; Vz -= wsum * piz;
    float ex = ux * Vx + uy * Vy + uz * Vz + 1e-5f;
    float ey = vx * Vx + vy * Vy + vz * Vz + 1e-5f;
    float t0x = ex * ux + ey * vx, t0y = ex * uy + ey * vy, t0z = ex * uz + ey * vz;
    float t1x = -ey * ux + ex * vx, t1y = -ey * uy + ex * vy, t1z = -ey * uz + ex * vz;
    float n0 = fmaxf(sqrtf(t0x * t0x + t0y * t0y + t0z * t0z), 1e-12f);
    float n1 = fmaxf(sqrtf(t1x * t1x + t1y * t1y + t1z * t1z), 1e-12f);
    float* o = nuv + (size_t)i * 9;
    o[0] = nx;       o[1] = ny;       o[2] = nz;
    o[3] = t0x / n0; o[4] = t0y / n0; o[5] = t0z / n0;
    o[6] = t1x / n1; o[7] = t1y / n1; o[8] = t1z / n1;
}

// ---------------------------------------------------------------------------
// generic WMMA dense: Y[N x 128] = act(X[N x K] @ Wf16[128 x K]^T + bias) (+resid)
// ---------------------------------------------------------------------------
template <int K>
__global__ __launch_bounds__(256)
void k_dense(const float* __restrict__ X, const _Float16* __restrict__ W,
             const float* __restrict__ bias, const float* __restrict__ resid,
             float* __restrict__ Y, int act) {
    const int i0 = blockIdx.x * 16;
    const int wave = threadIdx.x >> 5;
    const int lane = threadIdx.x & 31;
    const int mrow = lane & 15;
    const int hi   = lane >> 4;
    v8f acc = {};
    const int o = wave * 16 + mrow;
    const _Float16* wrow = W + (size_t)o * K;
    const float* xrow = X + (size_t)(i0 + mrow) * K;
    constexpr int NKC = (K + 31) / 32;
#pragma unroll
    for (int kc = 0; kc < NKC; ++kc) {
        const int kbase = kc * 32;
        const int K0 = kbase + (hi ? 8 : 0);
        v16h afrag;
        {
            float4 x0 = *(const float4*)(xrow + K0);
            float4 x1 = *(const float4*)(xrow + K0 + 4);
            afrag[0] = (_Float16)x0.x; afrag[1] = (_Float16)x0.y;
            afrag[2] = (_Float16)x0.z; afrag[3] = (_Float16)x0.w;
            afrag[4] = (_Float16)x1.x; afrag[5] = (_Float16)x1.y;
            afrag[6] = (_Float16)x1.z; afrag[7] = (_Float16)x1.w;
            if constexpr (K >= 32) {
                float4 x2 = *(const float4*)(xrow + K0 + 16);
                float4 x3 = *(const float4*)(xrow + K0 + 20);
                afrag[8]  = (_Float16)x2.x; afrag[9]  = (_Float16)x2.y;
                afrag[10] = (_Float16)x2.z; afrag[11] = (_Float16)x2.w;
                afrag[12] = (_Float16)x3.x; afrag[13] = (_Float16)x3.y;
                afrag[14] = (_Float16)x3.z; afrag[15] = (_Float16)x3.w;
            } else {
#pragma unroll
                for (int e = 8; e < 16; ++e) afrag[e] = (_Float16)0.f;
            }
        }
        v16h bfrag;
        if constexpr (K >= 32) {
            const int Kb0 = kbase + (hi ? 8 : 0);
            v8h b0 = *(const v8h*)(wrow + Kb0);
            v8h b1 = *(const v8h*)(wrow + Kb0 + 16);
            bfrag = __builtin_shufflevector(b0, b1, 0, 1, 2, 3, 4, 5, 6, 7, 8, 9, 10, 11, 12, 13, 14, 15);
        } else {
            v16h tmp = *(const v16h*)wrow;
#pragma unroll
            for (int e = 0; e < 16; ++e)
                bfrag[e] = (e < 8) ? tmp[(e & 7) + (hi ? 8 : 0)] : (_Float16)0.f;
        }
        acc = __builtin_amdgcn_wmma_f32_16x16x32_f16(false, afrag, false, bfrag,
                                                     (short)0, acc, false, false);
    }
    __syncthreads();   // allow X == Y in-place across waves of this tile
    const float bo = bias ? bias[o] : 0.f;
#pragma unroll
    for (int r = 0; r < 8; ++r) {
        int m = r + (hi ? 8 : 0);
        int i = i0 + m;
        if (i < NPTS) {
            float v = acc[r] + bo;
            if (act == 1)      v = (v >= 0.f) ? v : 0.2f * v;
            else if (act == 2) v = fmaxf(v, 0.f);
            if (resid) v += resid[(size_t)i * EMB + o];
            Y[(size_t)i * EMB + o] = v;
        }
    }
}

// ---------------------------------------------------------------------------
// group-norm
// ---------------------------------------------------------------------------
__global__ void k_gn_stats(const float* __restrict__ X, float* __restrict__ stats) {
    __shared__ float s[8];
    if (threadIdx.x < 8) s[threadIdx.x] = 0.f;
    __syncthreads();
    int c = threadIdx.x;
    int g = c >> 5;
    float sum = 0.f, sq = 0.f;
    for (int i = blockIdx.x; i < NPTS; i += gridDim.x) {
        float v = X[(size_t)i * EMB + c];
        sum += v; sq = fmaf(v, v, sq);
    }
    atomicAdd(&s[2 * g], sum);
    atomicAdd(&s[2 * g + 1], sq);
    __syncthreads();
    if (threadIdx.x < 8) atomicAdd(&stats[threadIdx.x], s[threadIdx.x]);
}

__global__ void k_gn_apply(const float* __restrict__ X, const float* __restrict__ stats,
                           const float* __restrict__ gamma, const float* __restrict__ beta,
                           float* __restrict__ Y, _Float16* __restrict__ hT) {
    int idx = blockIdx.x * blockDim.x + threadIdx.x;
    if (idx >= NPAD * EMB) return;
    int i = idx >> 7, c = idx & 127;
    if (i >= NPTS) {
        if (hT) hT[(size_t)c * NPAD + i] = (_Float16)0.f;
        return;
    }
    int g = c >> 5;
    const float cnt = (float)NPTS * 32.f;
    float mean = stats[2 * g] / cnt;
    float var  = stats[2 * g + 1] / cnt - mean * mean;
    float inv  = rsqrtf(var + 1e-5f);
    float v = (X[idx] - mean) * inv * gamma[c] + beta[c];
    Y[idx] = v;
    if (hT) hT[(size_t)c * NPAD + i] = (_Float16)v;
}

// ---------------------------------------------------------------------------
// TDM helper: async-load a 128-row x 32-half tile (row stride NPAD halves)
// from global (f16) into LDS at byte offset lds_off. D# per CDNA5 ISA ch.8.
// ---------------------------------------------------------------------------
#if HAVE_TDM
__device__ __forceinline__ void tdm_load_tile(const _Float16* gsrc, u32 lds_off) {
    unsigned long long ga = (unsigned long long)(size_t)gsrc;
    u32x4 g0;
    g0[0] = 1u;                                            // count=1, user-mode
    g0[1] = lds_off;                                       // lds_addr (bytes)
    g0[2] = (u32)(ga & 0xffffffffu);                       // global_addr[31:0]
    g0[3] = (u32)((ga >> 32) & 0x01ffffffu) | (2u << 30);  // addr[56:32], type=2
    i32x8 g1;
    g1[0] = 0x00010000;          // wg_mask=0, data_size=1 (2B), no flags
    g1[1] = (int)(3008u << 16);  // abar_addr=0 | tensor_dim0[15:0]=3008
    g1[2] = (int)(128u << 16);   // tensor_dim0 hi=0 | tensor_dim1[15:0]=128
    g1[3] = (int)(32u << 16);    // tensor_dim1 hi=0 | tile_dim0=32
    g1[4] = 128;                 // tile_dim1=128 | tile_dim2=0
    g1[5] = 3008;                // tensor_dim0_stride lo (48-bit)
    g1[6] = 0;                   // stride hi | dim1_stride lo
    g1[7] = 0;
    i32x4 z4 = {};
#if defined(__clang_major__) && (__clang_major__ >= 23)
    i32x8 z8 = {};
    __builtin_amdgcn_tensor_load_to_lds(g0, g1, z4, z4, z8, 0);
#else
    __builtin_amdgcn_tensor_load_to_lds(g0, g1, z4, z4, 0);
#endif
}
#endif

// ---------------------------------------------------------------------------
// pairwise conv: out[i,h] = sum_c A2[h,c] * T[i,c,h] + B2[h] * S[i,h]
// block = 288 threads (9 waves, wave c owns channel c), one 16-i tile / block
// h tile staged to LDS per K-chunk (TDM async DMA, double-buffered)
// ---------------------------------------------------------------------------
__global__ __launch_bounds__(288)
void k_conv(const float* __restrict__ xyz, const float* __restrict__ nrm,
            const float* __restrict__ nuv, const _Float16* __restrict__ hT,
            const float* __restrict__ A1, const float* __restrict__ B1,
            const float* __restrict__ A2, const float* __restrict__ B2,
            float* __restrict__ out) {
    __shared__ float    s_nuv[16][9];
    __shared__ float    s_pi[16][3];
    __shared__ float    s_A1[8][3];
    __shared__ float    s_B1[8];
    __shared__ _Float16 s_w[9][16][32];       // WMMA A source per cut
    __shared__ _Float16 s_hT[2][EMB][32];     // double-buffered h tile [chan][j]
    __shared__ float    s_out[16][EMB];

    const int i0   = blockIdx.x * 16;
    const int tid  = threadIdx.x;
    const int wave = tid >> 5;                // 0..8 == cut channel
    const int lane = tid & 31;
    const int mrow = lane & 15;
    const int hi   = lane >> 4;
    const float invSR = 0.078567420131839f;   // 1 / (sqrt(2) * RADIUS)

    // ---- preload per-tile constants ----
    if (tid < 144) {
        int ii = tid / 9, kk = tid % 9;
        s_nuv[ii][kk] = (i0 + ii < NPTS) ? nuv[(size_t)(i0 + ii) * 9 + kk] : 0.f;
    }
    if (tid >= 160 && tid < 208) {
        int t = tid - 160; int ii = t / 3, d = t % 3;
        s_pi[ii][d] = (i0 + ii < NPTS) ? xyz[(size_t)(i0 + ii) * 3 + d] * invSR : 0.f;
    }
    if (tid >= 224 && tid < 248) { int t = tid - 224; s_A1[t / 3][t % 3] = A1[t]; }
    if (tid >= 248 && tid < 256) s_B1[tid - 248] = B1[tid - 248];
    for (int e = tid; e < 16 * EMB; e += 288) ((float*)s_out)[e] = 0.f;

    // ---- stage chunk 0 ----
#if HAVE_TDM
    if (wave == 0)
        tdm_load_tile(hT, (u32)(size_t)&s_hT[0][0][0]);
#else
    for (int q = tid; q < 512; q += 288) {           // 512 x 16B
        int hch = q >> 2, part = q & 3;
        float4 v = *((const float4*)(hT + (size_t)hch * NPAD) + part);
        *((float4*)&s_hT[0][hch][0] + part) = v;
    }
#endif

    v8f acc[8] = {};

    for (int jc = 0; jc < NCHUNK; ++jc) {
        const int j0 = jc * 32;
        const int buf = jc & 1;
        __syncthreads();   // prev chunk's LDS reads complete; buffers reusable
        // ---- cooperative pairwise weight generation (512 pairs) ----
        for (int p = tid; p < 512; p += 288) {
            int il = p >> 5, jl = p & 31;
            int j = j0 + jl;
            float w9[9];
            if ((i0 + il) < NPTS && j < NPTS) {
                float pjx = xyz[j * 3 + 0] * invSR;
                float pjy = xyz[j * 3 + 1] * invSR;
                float pjz = xyz[j * 3 + 2] * invSR;
                float dx = pjx - s_pi[il][0];
                float dy = pjy - s_pi[il][1];
                float dz = pjz - s_pi[il][2];
                float cs = s_nuv[il][0] * nrm[j * 3 + 0] +
                           s_nuv[il][1] * nrm[j * 3 + 1] +
                           s_nuv[il][2] * nrm[j * 3 + 2];
                float t  = 2.f - cs;
                float d2 = (dx * dx + dy * dy + dz * dz) * t * t;
                float win = __expf(-d2);
                float X0 = s_nuv[il][0] * dx + s_nuv[il][1] * dy + s_nuv[il][2] * dz;
                float X1 = s_nuv[il][3] * dx + s_nuv[il][4] * dy + s_nuv[il][5] * dz;
                float X2 = s_nuv[il][6] * dx + s_nuv[il][7] * dy + s_nuv[il][8] * dz;
                w9[0] = win;
#pragma unroll
                for (int c = 0; c < CUTS; ++c) {
                    float g = fmaxf(s_A1[c][0] * X0 + s_A1[c][1] * X1 +
                                    s_A1[c][2] * X2 + s_B1[c], 0.f);
                    w9[c + 1] = win * g;
                }
            } else {
#pragma unroll
                for (int c = 0; c < 9; ++c) w9[c] = 0.f;
            }
#pragma unroll
            for (int c = 0; c < 9; ++c) s_w[c][il][jl] = (_Float16)w9[c];
        }
        // ---- stage next chunk into the other buffer / wait current tile ----
#if HAVE_TDM
        if (wave == 0) {
            if (jc + 1 < NCHUNK) {
                tdm_load_tile(hT + (size_t)(j0 + 32), (u32)(size_t)&s_hT[buf ^ 1][0][0]);
                __builtin_amdgcn_s_wait_tensorcnt(1);   // current tile landed
            } else {
                __builtin_amdgcn_s_wait_tensorcnt(0);
            }
        }
#else
        if (jc + 1 < NCHUNK) {
            for (int q = tid; q < 512; q += 288) {
                int hch = q >> 2, part = q & 3;
                float4 v = *((const float4*)(hT + (size_t)hch * NPAD + j0 + 32) + part);
                *((float4*)&s_hT[buf ^ 1][hch][0] + part) = v;
            }
        }
#endif
        __syncthreads();   // weights + staged tile visible to all waves
        // ---- A fragment for this wave's cut: two 16B LDS vector loads ----
        v16h afrag;
        {
            const v8h* rowp = (const v8h*)&s_w[wave][mrow][0];
            v8h lo  = rowp[2 * hi];        // halves K0..K0+7   (K0 = hi*16)
            v8h hi2 = rowp[2 * hi + 1];    // halves K0+8..K0+15
            afrag = __builtin_shufflevector(lo, hi2,
                                            0, 1, 2, 3, 4, 5, 6, 7,
                                            8, 9, 10, 11, 12, 13, 14, 15);
        }
        // ---- preload ALL B fragments (distinct regs -> clause + overlap) ----
        const int Kb0 = hi ? 16 : 0;
        v16h bfr[8];
#pragma unroll
        for (int nt = 0; nt < 8; ++nt)
            bfr[nt] = *(const v16h*)&s_hT[buf][nt * 16 + mrow][Kb0];
        // ---- 8 back-to-back WMMAs ----
#pragma unroll
        for (int nt = 0; nt < 8; ++nt)
            acc[nt] = __builtin_amdgcn_wmma_f32_16x16x32_f16(false, afrag, false, bfr[nt],
                                                             (short)0, acc[nt], false, false);
    }
    // ---- cross-cut reduction: scale per (h, c), accumulate via LDS atomics ----
#pragma unroll
    for (int nt = 0; nt < 8; ++nt) {
        int hch = nt * 16 + mrow;
        float coef = (wave == 0) ? B2[hch] : A2[(size_t)hch * CUTS + (wave - 1)];
#pragma unroll
        for (int r = 0; r < 8; ++r) {
            int m = r + (hi ? 8 : 0);
            atomicAdd(&s_out[m][hch], acc[nt][r] * coef);
        }
    }
    __syncthreads();
    for (int e = tid; e < 16 * EMB; e += 288) {
        int m = e >> 7, hch = e & 127;
        int i = i0 + m;
        if (i < NPTS) out[(size_t)i * EMB + hch] = ((float*)s_out)[e];
    }
}

// ---------------------------------------------------------------------------
// host-side orchestration
// ---------------------------------------------------------------------------
extern "C" void kernel_launch(void* const* d_in, const int* in_sizes, int n_in,
                              void* d_out, int out_size, void* d_ws, size_t ws_size,
                              hipStream_t stream) {
    (void)in_sizes; (void)n_in; (void)out_size; (void)ws_size;
    const float* feats = (const float*)d_in[IN_FEATS];
    const float* xyz   = (const float*)d_in[IN_XYZ];
    const float* nrm   = (const float*)d_in[IN_NRM];

    uint8_t* ws = (uint8_t*)d_ws;
    size_t off = 0;
    #define WALLOC(ty, name, cnt) ty* name = (ty*)(ws + off); \
        off = (off + (size_t)(cnt) * sizeof(ty) + 255) & ~(size_t)255;
    WALLOC(float, scores, NPAD)
    WALLOC(float, nuvbuf, NPAD * 9)
    WALLOC(float, featsP, NPAD * 16)
    WALLOC(float, X0, NPAD * EMB)
    WALLOC(float, Hb, NPAD * EMB)
    WALLOC(float, CO, NPAD * EMB)
    WALLOC(float, T1, NPAD * EMB)
    WALLOC(float, T2, NPAD * EMB)
    WALLOC(float, XI, NPAD * EMB)
    WALLOC(_Float16, hT16, EMB * NPAD)
    WALLOC(float, stats, 16)
    _Float16* wf16[3][7];
    for (int l = 0; l < 3; ++l)
        for (int m = 0; m < 7; ++m) { WALLOC(_Float16, wp, EMB * EMB) wf16[l][m] = wp; }
    #undef WALLOC

    // 1) orientation scores + tangent frames + padded features
    k_scores<<<(NPTS + 255) / 256, 256, 0, stream>>>(
        feats, (const float*)d_in[IN_ORIW1], (const float*)d_in[IN_ORIB1],
        (const float*)d_in[IN_ORIW2], (const float*)d_in[IN_ORIB2], scores);
    k_mesh<<<(NPTS + 255) / 256, 256, 0, stream>>>(xyz, nrm, scores, nuvbuf);
    k_padfeats<<<(NPAD * 16) / 256, 256, 0, stream>>>(feats, featsP);

    // 2) convert all layer weight matrices to f16 (row-major [out][in])
    const int widx[7] = {P_INW1, P_INW2, P_OUTW1, P_OUTW2, P_LINW1, P_LINW2, P_SKIPW};
    for (int l = 0; l < 3; ++l) {
        int inch = (l == 0) ? 16 : EMB;
        for (int m = 0; m < 7; ++m) {
            int K = (m == 0 || m == 6) ? inch : EMB;
            int n = EMB * K;
            k_cvt16<<<(n + 255) / 256, 256, 0, stream>>>(
                (const float*)d_in[LIDX(l, widx[m])], wf16[l][m], n);
        }
    }

    // 3) three conv layers
    for (int l = 0; l < 3; ++l) {
        const float* xin = (l == 0) ? featsP : X0;
        #define PL(k) ((const float*)d_in[LIDX(l, (k))])

        if (l == 0)
            k_dense<16 ><<<NTILE, 256, 0, stream>>>(xin, wf16[l][0], PL(P_INB1), nullptr, T1, 1);
        else
            k_dense<128><<<NTILE, 256, 0, stream>>>(xin, wf16[l][0], PL(P_INB1), nullptr, T1, 1);
        k_dense<128><<<NTILE, 256, 0, stream>>>(T1, wf16[l][1], PL(P_INB2), nullptr, T2, 1);
        k_zero<<<1, 32, 0, stream>>>(stats, 8);
        k_gn_stats<<<32, 128, 0, stream>>>(T2, stats);
        k_gn_apply<<<(NPAD * EMB) / 256, 256, 0, stream>>>(
            T2, stats, PL(P_GNING), PL(P_GNINB), Hb, hT16);

        k_conv<<<NTILE, 288, 0, stream>>>(xyz, nrm, nuvbuf, hT16,
                                          PL(P_A1), PL(P_B1), PL(P_A2), PL(P_B2), CO);

        k_dense<128><<<NTILE, 256, 0, stream>>>(CO, wf16[l][2], PL(P_OUTB1), nullptr, T1, 1);
        k_dense<128><<<NTILE, 256, 0, stream>>>(T1, wf16[l][3], PL(P_OUTB2), nullptr, T2, 0);
        k_zero<<<1, 32, 0, stream>>>(stats, 8);
        k_gn_stats<<<32, 128, 0, stream>>>(T2, stats);
        k_gn_apply<<<(NPAD * EMB) / 256, 256, 0, stream>>>(
            T2, stats, PL(P_GNOUTG), PL(P_GNOUTB), CO, (_Float16*)nullptr);

        k_dense<128><<<NTILE, 256, 0, stream>>>(CO, wf16[l][4], PL(P_LINB1), nullptr, T1, 2);
        k_dense<128><<<NTILE, 256, 0, stream>>>(T1, wf16[l][5], PL(P_LINB2), nullptr, XI, 0);
        float* ydst = (l == 2) ? (float*)d_out : X0;
        if (l == 0)
            k_dense<16 ><<<NTILE, 256, 0, stream>>>(xin, wf16[l][6], PL(P_SKIPB), XI, ydst, 0);
        else
            k_dense<128><<<NTILE, 256, 0, stream>>>(xin, wf16[l][6], PL(P_SKIPB), XI, ydst, 0);
        #undef PL
    }
}
